// myVMLSTM_Group_21998822490487
// MI455X (gfx1250) — compile-verified, matrix operand order checked by reading, not run on hardware
//
#include <hip/hip_runtime.h>

typedef __attribute__((ext_vector_type(16))) __bf16 v16bf;
typedef __attribute__((ext_vector_type(8)))  float  v8f;

#define T_   512
#define B_   40
#define D_   1024
#define H_   1024
#define WR_  256
#define TB_  (T_ * B_)      // 20480
#define FH_  4096
#define NWG  32

// ---------------- WMMA helpers (CDNA5 16x16x32 bf16, f32 accumulate) ----------------

__device__ __forceinline__ v8f wmma32(v16bf a, v16bf b, v8f c) {
  // 8 args: (neg_a, A, neg_b, B, c_mod, C, reuse_a, reuse_b)
  return __builtin_amdgcn_wmma_f32_16x16x32_bf16(false, a, false, b, (short)0, c, false, false);
}

// K index of fragment element e for this lane's K-half (lane>>4), per ISA 16-bit A layout:
// VGPR0..3 hold K = khalf*8 + 0..7, VGPR4..7 hold K = 16 + khalf*8 + 0..7
__device__ __forceinline__ int frag_k(int e, int kh) {
  return ((e >> 3) << 4) + kh * 8 + (e & 7);
}

// Fragment with contiguous-K rows: element = p[row*ld + k], row = lane&15.
// Used for A (M-major) and for B operands stored N-major. Each lane-half reads
// two contiguous 16-byte runs -> coalesces into b128 loads.
__device__ __forceinline__ v16bf frag_rowk_bf(const __bf16* p, int ld, int lane) {
  int row = lane & 15, kh = lane >> 4;
  v16bf f;
#pragma unroll
  for (int e = 0; e < 16; ++e) f[e] = p[row * ld + frag_k(e, kh)];
  return f;
}

__device__ __forceinline__ v16bf frag_rowk_bf_guard(const __bf16* p, int ld, int lane,
                                                    int rowbase, int rowmax) {
  int row = lane & 15, kh = lane >> 4;
  bool ok = (rowbase + row) < rowmax;
  v16bf f;
#pragma unroll
  for (int e = 0; e < 16; ++e) f[e] = ok ? p[row * ld + frag_k(e, kh)] : (__bf16)0.0f;
  return f;
}

// ---------------- device-scope barrier for the persistent recurrence kernel ----------------

__device__ __forceinline__ void gbar(unsigned* cnt, unsigned* gen, unsigned nb) {
  __syncthreads();
  if (threadIdx.x == 0) {
    __threadfence();
    unsigned g = __hip_atomic_load(gen, __ATOMIC_RELAXED, __HIP_MEMORY_SCOPE_AGENT);
    unsigned prev = __hip_atomic_fetch_add(cnt, 1u, __ATOMIC_ACQ_REL, __HIP_MEMORY_SCOPE_AGENT);
    if (prev == nb - 1u) {
      __hip_atomic_store(cnt, 0u, __ATOMIC_RELAXED, __HIP_MEMORY_SCOPE_AGENT);
      __hip_atomic_store(gen, g + 1u, __ATOMIC_RELEASE, __HIP_MEMORY_SCOPE_AGENT);
    } else {
      while (__hip_atomic_load(gen, __ATOMIC_ACQUIRE, __HIP_MEMORY_SCOPE_AGENT) == g)
        __builtin_amdgcn_s_sleep(1);
    }
    __threadfence();
  }
  __syncthreads();
}

// ---------------- phase 0a: one-shot x -> bf16 (removes 16x redundant cvt from gemm1) ----------

__global__ void xconv_kernel(const float* __restrict__ x, __bf16* __restrict__ xbf) {
  size_t idx = (size_t)blockIdx.x * blockDim.x + threadIdx.x;   // TB_*D_ / 4 threads
  size_t base = idx * 4;
  float4 v = *(const float4*)(x + base);
  __bf16 o0 = (__bf16)v.x, o1 = (__bf16)v.y, o2 = (__bf16)v.z, o3 = (__bf16)v.w;
  xbf[base + 0] = o0;
  xbf[base + 1] = o1;
  xbf[base + 2] = o2;
  xbf[base + 3] = o3;
}

// ---------------- phase 0b: weight repack (N-major, bf16) + diagonal hoisting + state init --------

__global__ void prep_kernel(const float* u_x, const float* w_x,
                            const float* u_h0, const float* V_h0,
                            const float* u_h1, const float* V_h1,
                            const float* dia_x, const float* dia_h,
                            const float* h0, const float* c0,
                            __bf16* uxT, __bf16* wxb, __bf16* UcatT, __bf16* VcatT,
                            float* axd, float* ahd,
                            float* h, float* cst, __bf16* hbf, unsigned* bar) {
  int idx = blockIdx.x * blockDim.x + threadIdx.x;
  // uxT[n][k] = u_x[k][n]   (256 x 1024, N-major)
  if (idx < WR_ * D_) {
    int n = idx >> 10, k = idx & 1023;
    uxT[idx] = (__bf16)u_x[k * WR_ + n];
  }
  // wxb = w_x as-is (4096 x 256 is already N-major for B = w_x^T)
  if (idx < FH_ * WR_) wxb[idx] = (__bf16)w_x[idx];
  // UcatT[p][g][r][k] = u_h{p}[g][k][r]   (per (p,g): 64 x 512, N-major)
  if (idx < 2 * 2 * 64 * 512) {
    int p = idx >> 16;
    int rem = idx & 65535;
    int g = rem >> 15;
    int r = (rem >> 9) & 63;
    int k = rem & 511;
    const float* src = (p == 0) ? u_h0 : u_h1;
    UcatT[idx] = (__bf16)src[g * 32768 + k * 64 + r];
  }
  // VcatT[g][m][k] : k<64 -> V_h0[g][k][m], k>=64 -> V_h1[g][k-64][m]   (per g: 2048 x 128)
  if (idx < 2 * 2048 * 128) {
    int g = idx >> 18;
    int m = (idx >> 7) & 2047;
    int k = idx & 127;
    float v = (k < 64) ? V_h0[g * 131072 + k * 2048 + m]
                       : V_h1[g * 131072 + (k - 64) * 2048 + m];
    VcatT[idx] = (__bf16)v;
  }
  if (idx < FH_) {
    int i = idx & 1023;
    // axd[j] = dia_x[i] - sum_r u_x[i,r] * w_x[j,r]
    float s = 0.f;
    for (int r = 0; r < WR_; ++r) s += u_x[i * WR_ + r] * w_x[idx * WR_ + r];
    axd[idx] = dia_x[i] - s;
    // ahd[j] = dia_h[i] - diag_h[j>>10, i];  diag_h from u_h0 / V_h0 only
    int g = idx >> 11;
    int m = idx & 2047;
    float s2 = 0.f;
    for (int r = 0; r < 64; ++r) s2 += u_h0[i * 64 + r] * V_h0[g * 131072 + r * 2048 + m];
    ahd[idx] = dia_h[i] - s2;
  }
  if (idx < B_ * H_) {
    h[idx] = h0[idx];
    cst[idx] = c0[idx];
    hbf[idx] = (__bf16)h0[idx];
  }
  if (idx < 2) bar[idx] = 0u;
}

// ---------------- phase 1: L = bf16( X(20480x1024) @ u_x(1024x256) ), pure bf16 WMMA ----------------

__global__ void gemm1_kernel(const __bf16* __restrict__ xbf, const __bf16* __restrict__ uxT,
                             __bf16* __restrict__ L) {
  int gw = (blockIdx.x * blockDim.x + threadIdx.x) >> 5;   // one wave per 16x16 tile
  int lane = threadIdx.x & 31;
  int mt = gw >> 4;        // 1280 row tiles
  int nt = gw & 15;        // 16 col tiles
  const __bf16* Ab = xbf + (size_t)mt * 16 * D_;
  const __bf16* Bb = uxT + (size_t)nt * 16 * D_;   // N-major: row = n, contiguous k
  v8f acc = {};
  for (int k0 = 0; k0 < D_; k0 += 32) {
    v16bf a = frag_rowk_bf(Ab + k0, D_, lane);
    v16bf b = frag_rowk_bf(Bb + k0, D_, lane);
    acc = wmma32(a, b, acc);
  }
  int n = lane & 15, half = lane >> 4;
#pragma unroll
  for (int r = 0; r < 8; ++r) {
    int m = mt * 16 + r + 8 * half;
    L[(size_t)m * WR_ + nt * 16 + n] = (__bf16)acc[r];
  }
}

// ---------------- phase 2: Gx = L @ w_x^T + x*(dia_x - diag_x) + b_x  (fused epilogue) ----------------

__global__ void gemm2_kernel(const float* __restrict__ x, const __bf16* __restrict__ L,
                             const __bf16* __restrict__ wxb, const float* __restrict__ axd,
                             const float* __restrict__ b_x, float* __restrict__ Gx) {
  int gw = (blockIdx.x * blockDim.x + threadIdx.x) >> 5;
  int lane = threadIdx.x & 31;
  int mt = gw >> 8;        // 1280 row tiles
  int ct = gw & 255;       // 256 col tiles (4096/16)
  const __bf16* Ab = L + (size_t)mt * 16 * WR_;
  const __bf16* Bb = wxb + (size_t)ct * 16 * WR_;   // N-major: B[k][n] = w_x[n_global, k]
  v8f acc = {};
  for (int k0 = 0; k0 < WR_; k0 += 32) {
    v16bf a = frag_rowk_bf(Ab + k0, WR_, lane);
    v16bf b = frag_rowk_bf(Bb + k0, WR_, lane);
    acc = wmma32(a, b, acc);
  }
  int n = lane & 15, half = lane >> 4;
  int j = ct * 16 + n;
  int i = j & 1023;
  float ax = axd[j], bx = b_x[j];
#pragma unroll
  for (int r = 0; r < 8; ++r) {
    int m = mt * 16 + r + 8 * half;
    float v = acc[r] + x[(size_t)m * D_ + i] * ax + bx;
    __builtin_nontemporal_store(v, &Gx[(size_t)m * FH_ + j]);   // streamed, don't pollute L2
  }
}

// ---------------- phase 3: persistent recurrence (2 device barriers / step) ----------------

__global__ void __launch_bounds__(256, 1)
rec_kernel(const float* __restrict__ Gx, const __bf16* __restrict__ UcatT,
           const __bf16* __restrict__ VcatT, const float* __restrict__ ahd,
           const float* __restrict__ b_h,
           float* h, float* cst, __bf16* hbf, __bf16* abuf,
           unsigned* bar, float* out) {
  int gw = (blockIdx.x * 256 + threadIdx.x) >> 5;   // 256 waves total
  int lane = threadIdx.x & 31;
  unsigned* cnt = bar;
  unsigned* gen = bar + 1;

  for (int t = 0; t < T_; ++t) {
    // ---- stage 1: a[p][g] = h[:, ((g+p)&1)*512 : +512] @ U_p[g]  (40x512 @ 512x64) ----
    if (gw < 48) {
      int pg = gw / 12;                 // (p,g) pair
      int rem = gw % 12;
      int mt = rem >> 2;                // 3 row tiles (rows 0..47, guarded at 40)
      int nt = rem & 3;                 // 4 col tiles of R=64
      int p = pg >> 1, g = pg & 1;
      int hoff = ((g + p) & 1) * 512;   // cumulative group rotation
      const __bf16* Ab = hbf + (size_t)mt * 16 * H_ + hoff;
      const __bf16* Bb = UcatT + ((size_t)pg * 64 + nt * 16) * 512;  // N-major rows of 512
      v8f acc = {};
      for (int k0 = 0; k0 < 512; k0 += 32) {
        v16bf a = frag_rowk_bf_guard(Ab + k0, H_, lane, mt * 16, B_);
        v16bf b = frag_rowk_bf(Bb + k0, 512, lane);
        acc = wmma32(a, b, acc);
      }
      int n = lane & 15, half = lane >> 4;
#pragma unroll
      for (int r = 0; r < 8; ++r) {
        int b = mt * 16 + r + 8 * half; // rows >=40 are zeros (guarded A), safe to store
        abuf[(size_t)g * 48 * 128 + b * 128 + p * 64 + nt * 16 + n] = (__bf16)acc[r];
      }
    }
    gbar(cnt, gen, NWG);

    // ---- stage 2 + gate epilogue: one wave owns a quad of coupled gate tiles ----
    if (gw < 192) {
      int it = gw / 3;                  // 64 i-tiles of 16 hidden columns
      int mt = gw % 3;                  // 3 row tiles
      v8f acc[4];
#pragma unroll
      for (int c = 0; c < 4; ++c) {     // chunks i,f,o,n  ->  (g = c>>1, m = (c&1)*1024 + i)
        int g = c >> 1;
        int mcol = ((c & 1) << 10) + it * 16;
        const __bf16* Ab = abuf + (size_t)g * 48 * 128 + mt * 16 * 128;
        const __bf16* Bb = VcatT + ((size_t)g * 2048 + mcol) * 128;  // N-major rows of 128
        v8f a2 = {};
        for (int k0 = 0; k0 < 128; k0 += 32) {
          v16bf af = frag_rowk_bf(Ab + k0, 128, lane);
          v16bf bf = frag_rowk_bf(Bb + k0, 128, lane);
          a2 = wmma32(af, bf, a2);
        }
        acc[c] = a2;
      }
      int n = lane & 15, half = lane >> 4;
      int i = it * 16 + n;
      float ah0 = ahd[i], ah1 = ahd[1024 + i], ah2 = ahd[2048 + i], ah3 = ahd[3072 + i];
      float bh0 = b_h[i], bh1 = b_h[1024 + i], bh2 = b_h[2048 + i], bh3 = b_h[3072 + i];
#pragma unroll
      for (int r = 0; r < 8; ++r) {
        int b = mt * 16 + r + 8 * half;
        if (b < B_) {
          float hv = h[(size_t)b * H_ + i];
          const float* gxr = Gx + ((size_t)t * B_ + b) * FH_;
          float g0 = gxr[i]        + acc[0][r] + hv * ah0 + bh0;
          float g1 = gxr[1024 + i] + acc[1][r] + hv * ah1 + bh1;
          float g2 = gxr[2048 + i] + acc[2][r] + hv * ah2 + bh2;
          float g3 = gxr[3072 + i] + acc[3][r] + hv * ah3 + bh3;
          float ig = 1.f / (1.f + __expf(-g0));
          float fg = 1.f / (1.f + __expf(-g1));
          float og = 1.f / (1.f + __expf(-g2));
          float ng = tanhf(g3);
          float cv = cst[(size_t)b * H_ + i];
          float c2 = fg * cv + ig * ng;
          float h2 = og * tanhf(c2);
          cst[(size_t)b * H_ + i] = c2;
          h[(size_t)b * H_ + i] = h2;
          hbf[(size_t)b * H_ + i] = (__bf16)h2;
          __builtin_nontemporal_store(h2, &out[((size_t)t * B_ + b) * H_ + i]);
          if (t == T_ - 1) {
            out[(size_t)T_ * B_ * H_ + (size_t)b * H_ + i] = h2;               // hT
            out[(size_t)T_ * B_ * H_ + B_ * H_ + (size_t)b * H_ + i] = c2;    // cT
          }
        }
      }
    }
    gbar(cnt, gen, NWG);
  }
}

// ---------------- host launcher ----------------

extern "C" void kernel_launch(void* const* d_in, const int* in_sizes, int n_in,
                              void* d_out, int out_size, void* d_ws, size_t ws_size,
                              hipStream_t stream) {
  const float* x     = (const float*)d_in[0];
  const float* h0    = (const float*)d_in[1];
  const float* c0    = (const float*)d_in[2];
  const float* u_x   = (const float*)d_in[3];
  const float* w_x   = (const float*)d_in[4];
  const float* u_h0  = (const float*)d_in[5];
  const float* V_h0  = (const float*)d_in[6];
  const float* u_h1  = (const float*)d_in[7];
  const float* V_h1  = (const float*)d_in[8];
  const float* b_x   = (const float*)d_in[9];
  const float* b_h   = (const float*)d_in[10];
  const float* dia_x = (const float*)d_in[11];
  const float* dia_h = (const float*)d_in[12];
  (void)in_sizes; (void)n_in; (void)out_size; (void)ws_size;

  char* ws = (char*)d_ws;
  size_t o = 0;
  auto carve = [&](size_t bytes) -> void* {
    void* p = ws + o;
    o += (bytes + 255) & ~(size_t)255;
    return p;
  };
  float*    Gx    = (float*)   carve((size_t)TB_ * FH_ * 4);     // 335.5 MB
  __bf16*   xbf   = (__bf16*)  carve((size_t)TB_ * D_ * 2);      // 40 MB
  __bf16*   L     = (__bf16*)  carve((size_t)TB_ * WR_ * 2);     // 10.5 MB
  __bf16*   uxT   = (__bf16*)  carve((size_t)WR_ * D_ * 2);
  __bf16*   wxb   = (__bf16*)  carve((size_t)FH_ * WR_ * 2);
  __bf16*   UcatT = (__bf16*)  carve((size_t)2 * 2 * 64 * 512 * 2);
  __bf16*   VcatT = (__bf16*)  carve((size_t)2 * 2048 * 128 * 2);
  float*    axd   = (float*)   carve((size_t)FH_ * 4);
  float*    ahd   = (float*)   carve((size_t)FH_ * 4);
  float*    hbuf  = (float*)   carve((size_t)B_ * H_ * 4);
  float*    cbuf  = (float*)   carve((size_t)B_ * H_ * 4);
  __bf16*   hbf   = (__bf16*)  carve((size_t)B_ * H_ * 2);
  __bf16*   abuf  = (__bf16*)  carve((size_t)2 * 48 * 128 * 2);
  unsigned* bar   = (unsigned*)carve(256);

  xconv_kernel<<<TB_ * D_ / 4 / 256, 256, 0, stream>>>(x, xbf);  // 20480 blocks
  prep_kernel<<<4096, 256, 0, stream>>>(u_x, w_x, u_h0, V_h0, u_h1, V_h1,
                                        dia_x, dia_h, h0, c0,
                                        uxT, wxb, UcatT, VcatT, axd, ahd,
                                        hbuf, cbuf, hbf, bar);
  gemm1_kernel<<<2560, 256, 0, stream>>>(xbf, uxT, L);           // 20480 tiles, 1 wave each
  gemm2_kernel<<<40960, 256, 0, stream>>>(x, L, wxb, axd, b_x, Gx); // 327680 tiles
  rec_kernel<<<NWG, 256, 0, stream>>>(Gx, UcatT, VcatT, ahd, b_h,
                                      hbuf, cbuf, hbf, abuf, bar, (float*)d_out);
}
